// LocalMultiHeadChannelAttention_57286273794418
// MI455X (gfx1250) — compile-verified, matrix operand order checked by reading, not run on hardware
//
#include <hip/hip_runtime.h>

// LocalMultiHeadChannelAttention for MI455X (gfx1250, wave32, WMMA bf16->f32).
// B=16, C=512, R=32, PS=3, HN=8, D=128, H=W=96.

#define Bn  16
#define Cc  512
#define RR  1024
#define HNh 8
#define Dd  128
#define HW  96
#define PB  524288   // per-batch pooled elements C*R*R

typedef unsigned short u16;
typedef unsigned int   u32;
typedef __attribute__((ext_vector_type(16))) __bf16 v16bf;
typedef __attribute__((ext_vector_type(8)))  __bf16 v8bf;
typedef __attribute__((ext_vector_type(8)))  float  v8f;
typedef __attribute__((ext_vector_type(4)))  int    i32x4;

// gfx1250 async global->LDS path (guarded; falls back to manual copy)
#if defined(__has_builtin)
#if __has_builtin(__builtin_amdgcn_global_load_async_to_lds_b128) && \
    __has_builtin(__builtin_amdgcn_s_wait_asynccnt)
#define USE_ASYNC_LDS 1
#endif
#endif

#define AS1P __attribute__((address_space(1))) i32x4*
#define AS3P __attribute__((address_space(3))) i32x4*

struct bfpair { v8bf lo; v8bf hi; };

static __device__ __forceinline__ v16bf joinbf(v8bf lo, v8bf hi) {
  bfpair p{lo, hi};
  return __builtin_bit_cast(v16bf, p);
}
static __device__ __forceinline__ u16 f2bf(float f) {
  u32 u = __builtin_bit_cast(u32, f);
  u32 r = u + 0x7fffu + ((u >> 16) & 1u);   // round-to-nearest-even
  return (u16)(r >> 16);
}
static __device__ __forceinline__ v8bf ld8(const u16* p) { return *(const v8bf*)p; }

static __device__ __forceinline__ v8f wmma_bf16(v16bf a, v16bf b, v8f c) {
  // D(f32 16x16) = A(bf16 16x32) * B(bf16 32x16) + C
  return __builtin_amdgcn_wmma_f32_16x16x32_bf16(false, a, false, b, (short)0, c,
                                                 false, false);
}

// ---------------------------------------------------------------- pooling
// avg/max 3x3 pool; write f32 avg (residual + V source) and bf16 avg/max.
__global__ void pool_kernel(const float* __restrict__ x, float* __restrict__ qpf,
                            u16* __restrict__ qpb, u16* __restrict__ kpb) {
  int idx = blockIdx.x * 256 + threadIdx.x;           // < Bn*PB exactly
  int b = idx >> 19;
  int rem = idx & (PB - 1);
  int c = rem >> 10;
  int t = rem & 1023;
  int r1 = t >> 5, r2 = t & 31;
  const float* base = x + ((b * Cc + c) * HW + r1 * 3) * HW + r2 * 3;
  float s = 0.f, mx = -3.4e38f;
  #pragma unroll
  for (int i = 0; i < 3; ++i)
    #pragma unroll
    for (int j = 0; j < 3; ++j) {
      float v = base[i * HW + j];
      s += v; mx = fmaxf(mx, v);
    }
  s *= (1.f / 9.f);
  qpf[idx] = s;
  qpb[idx] = f2bf(s);
  kpb[idx] = f2bf(mx);
}

// ---------------------------------------------------------------- weight cvt
__global__ void cvt_kernel(const float* __restrict__ wqk, const float* __restrict__ wv,
                           u16* __restrict__ wqkb, u16* __restrict__ wvb) {
  int i = blockIdx.x * 256 + threadIdx.x;             // 393216 exactly
  if (i < HNh * Dd * Dd) wqkb[i] = f2bf(wqk[i]);
  int j = i - HNh * Dd * Dd;
  if (j >= 0) wvb[j] = f2bf(wv[j]);
}

// ---------------------------------------------------------------- q/k proj
// out[b,h,c,dp] = sum_d heads(pool)[c,d] * Wqk[h,dp,d] + bqk[h,dp]
// heads(pool)[c,d] = pool_flat[(h*128+d)*512 + c]  (column-major -> LDS transpose)
__global__ void qkproj_kernel(const u16* __restrict__ poolb, const u16* __restrict__ wqkb,
                              const float* __restrict__ bqk, u16* __restrict__ out) {
  __shared__ u16 ldsA[128 * 136];                     // [cLocal][d], +8 pad
  int b = blockIdx.z, h = blockIdx.y, c0 = blockIdx.x * 128;
  int tid = threadIdx.x;
  const u32* src = (const u32*)(poolb + b * PB + h * Dd * Cc + c0);
  for (int i = tid; i < 128 * 64; i += 256) {         // 128 d x 128 c (pairs)
    int d = i >> 6, cp = i & 63;
    u32 v = src[d * 256 + cp];
    ldsA[(2 * cp + 0) * 136 + d] = (u16)(v & 0xffff);
    ldsA[(2 * cp + 1) * 136 + d] = (u16)(v >> 16);
  }
  __syncthreads();
  int wave = tid >> 5, lane = tid & 31;
  int row = wave * 16 + (lane & 15);                  // M row (c)
  int k0 = (lane >> 4) * 8;
  u16* op = out + (size_t)(b * HNh + h) * Cc * Dd;
  for (int dpt = 0; dpt < 8; ++dpt) {
    int dp = dpt * 16 + (lane & 15);                  // N col
    const u16* wrow = wqkb + h * (Dd * Dd) + dp * Dd;
    v8f acc = {};
    #pragma unroll
    for (int kb = 0; kb < 128; kb += 32) {
      v16bf A  = joinbf(ld8(&ldsA[row * 136 + kb + k0]),
                        ld8(&ldsA[row * 136 + kb + k0 + 16]));
      v16bf Bm = joinbf(ld8(wrow + kb + k0), ld8(wrow + kb + k0 + 16));
      acc = wmma_bf16(A, Bm, acc);
    }
    float bias = bqk[h * Dd + dp];
    #pragma unroll
    for (int g = 0; g < 8; ++g) {
      int c = c0 + wave * 16 + g + 8 * (lane >> 4);
      op[c * Dd + dp] = f2bf(acc[g] + bias);
    }
  }
}

// ---------------------------------------------------------------- V GEMM
// vpool[b,o,rr] = sum_c Wv[o,c]*qpool[b,c,rr] + bv[o]; stored flat o*1024+rr,
// which IS the VT[b,h,d,e] layout the attention kernel consumes.
__global__ void vgemm_kernel(const u16* __restrict__ poolb, const u16* __restrict__ wvb,
                             const float* __restrict__ bv, u16* __restrict__ vout) {
  __shared__ u16 ldsB[128 * 136];                     // [rrLocal][cChunk]
  int b = blockIdx.z;
  int o0 = blockIdx.y * 128, rr0 = blockIdx.x * 128;
  int tid = threadIdx.x, wave = tid >> 5, lane = tid & 31;
  int k0 = (lane >> 4) * 8;
  int orow = o0 + wave * 16 + (lane & 15);
  v8f zero = {};
  v8f acc[8];
  #pragma unroll
  for (int r = 0; r < 8; ++r) acc[r] = zero;
  const u32* src = (const u32*)(poolb + b * PB + rr0);
  for (int kc = 0; kc < 4; ++kc) {
    int cb = kc * 128;
    __syncthreads();
    for (int i = tid; i < 128 * 64; i += 256) {       // transpose-stage into LDS
      int cc = i >> 6, rp = i & 63;
      u32 v = src[(cb + cc) * 512 + rp];
      ldsB[(2 * rp + 0) * 136 + cc] = (u16)(v & 0xffff);
      ldsB[(2 * rp + 1) * 136 + cc] = (u16)(v >> 16);
    }
    if (kc < 3)                                       // warm L2 for next K chunk
      __builtin_prefetch((const void*)(src + (cb + 128 + (tid & 127)) * 512), 0, 3);
    __syncthreads();
    const u16* arow = wvb + orow * Cc + cb;
    #pragma unroll
    for (int kb = 0; kb < 128; kb += 32) {
      v16bf A = joinbf(ld8(arow + kb + k0), ld8(arow + kb + k0 + 16));
      #pragma unroll
      for (int rrt = 0; rrt < 8; ++rrt) {
        int rl = rrt * 16 + (lane & 15);
        v16bf Bm = joinbf(ld8(&ldsB[rl * 136 + kb + k0]),
                          ld8(&ldsB[rl * 136 + kb + k0 + 16]));
        acc[rrt] = wmma_bf16(A, Bm, acc[rrt]);
      }
    }
  }
  #pragma unroll
  for (int rrt = 0; rrt < 8; ++rrt)
    #pragma unroll
    for (int g = 0; g < 8; ++g) {
      int o  = o0 + wave * 16 + g + 8 * (lane >> 4);
      int rr = rr0 + rrt * 16 + (lane & 15);
      vout[b * PB + o * RR + rr] = f2bf(acc[rrt][g] + bv[o]);
    }
}

// ---------------------------------------------------------------- scores
// S[bh,c,e] = sum_d Q[bh,c,d]*K[bh,e,d].
// Block = 64(c) x 128(e) tile: K tile staged in LDS (async global->LDS when
// available) and shared by all 8 waves; each wave keeps its A fragment in
// registers across 4 e-subtiles.
__global__ void scores_kernel(const u16* __restrict__ Q, const u16* __restrict__ K,
                              float* __restrict__ S) {
  __shared__ u16 ldsK[128 * 136];                     // [eLocal][d], +8 pad
  int bh = blockIdx.z;
  int c0 = blockIdx.y * 64;
  int e0 = blockIdx.x * 128;
  int tid = threadIdx.x, wave = tid >> 5, lane = tid & 31;
  const u16* kbase = K + (size_t)bh * Cc * Dd + (size_t)e0 * Dd;
#if USE_ASYNC_LDS
  for (int task = tid; task < 128 * 16; task += 256) { // 128 rows x 16B chunks
    int row = task >> 4, ch = task & 15;
    __builtin_amdgcn_global_load_async_to_lds_b128(
        (AS1P)(kbase + row * Dd + ch * 8),
        (AS3P)&ldsK[row * 136 + ch * 8],
        0, 0);
  }
  __builtin_amdgcn_s_wait_asynccnt(0);
  __syncthreads();
#else
  for (int i = tid; i < 128 * 64; i += 256) {          // natural-layout copy
    int row = i >> 6, dp2 = i & 63;
    u32 v = ((const u32*)kbase)[row * 64 + dp2];
    *(u32*)&ldsK[row * 136 + dp2 * 2] = v;
  }
  __syncthreads();
#endif
  int csub = wave & 3;                                 // 4 c-subtiles of 16
  int ebase = (wave >> 2) * 4;                         // 4 e-subtiles per wave
  int k0 = (lane >> 4) * 8;
  const u16* qrow = Q + (size_t)bh * Cc * Dd + (c0 + csub * 16 + (lane & 15)) * Dd;
  v8f zero = {};
  v8f acc[4];
  #pragma unroll
  for (int et = 0; et < 4; ++et) acc[et] = zero;
  #pragma unroll
  for (int kb = 0; kb < 128; kb += 32) {
    v16bf A = joinbf(ld8(qrow + kb + k0), ld8(qrow + kb + k0 + 16));
    #pragma unroll
    for (int et = 0; et < 4; ++et) {
      int eL = (ebase + et) * 16 + (lane & 15);
      v16bf Bm = joinbf(ld8(&ldsK[eL * 136 + kb + k0]),
                        ld8(&ldsK[eL * 136 + kb + k0 + 16]));
      acc[et] = wmma_bf16(A, Bm, acc[et]);
    }
  }
  float* so = S + (size_t)bh * (Cc * Cc);
  #pragma unroll
  for (int et = 0; et < 4; ++et)
    #pragma unroll
    for (int g = 0; g < 8; ++g) {
      int c = c0 + csub * 16 + g + 8 * (lane >> 4);
      int e = e0 + (ebase + et) * 16 + (lane & 15);
      so[c * Cc + e] = acc[et][g];
    }
}

// ---------------------------------------------------------------- gate
// p[bh,c] = sigmoid( sum_cc Wp[c,cc]*mean_e S[bh,cc,e] + bp[c] )
__global__ void gate_kernel(const float* __restrict__ S, const float* __restrict__ Wp,
                            const float* __restrict__ bp, float* __restrict__ pgate) {
  __shared__ float m[512];
  int bh = blockIdx.x, tid = threadIdx.x;
  const float* sb = S + (size_t)bh * (Cc * Cc);
  int j = tid & 15;
  for (int base = 0; base < 512; base += 16) {
    int r = base + (tid >> 4);
    float v = 0.f;
    #pragma unroll 4
    for (int k = 0; k < 32; ++k) v += sb[r * Cc + j + 16 * k];
    #pragma unroll
    for (int o = 8; o > 0; o >>= 1) v += __shfl_down(v, o, 16);
    if (j == 0) m[r] = v * (1.f / 512.f);
  }
  __syncthreads();
  for (int c = tid; c < 512; c += 256) {
    float dot = bp[c];
    const float* wr = Wp + c * Cc;
    for (int cc = 0; cc < 512; ++cc) dot += wr[cc] * m[cc];
    pgate[bh * Cc + c] = 1.f / (1.f + __expf(-dot));
  }
}

// ---------------------------------------------------------------- softmax+attn+out
// w = softmax_e( S[c,e] * 128^-(0.5+p[c]) );  attn[c,d] = sum_e w[c,e]*v[e,d]
// out_flat[b, (h*128+d)*512 + c] = qpool_flat[same] + attn*(1+weight)
__global__ void attn_kernel(const float* __restrict__ S, const float* __restrict__ pgate,
                            const u16* __restrict__ V, const float* __restrict__ qpf,
                            const int* __restrict__ weight, float* __restrict__ out) {
  __shared__ u16 wlds[16 * 520];                      // [row c][e], +8 pad
  int c0 = blockIdx.x * 16, h = blockIdx.y, b = blockIdx.z;
  int bh = b * HNh + h;
  int tid = threadIdx.x, wave = tid >> 5, lane = tid & 31;
  const float* sb = S + (size_t)bh * (Cc * Cc);
  #pragma unroll
  for (int rr = 0; rr < 2; ++rr) {                    // 8 waves x 2 rows = 16 rows
    int ri = wave * 2 + rr;
    int c = c0 + ri;
    float invs = exp2f(-7.f * (0.5f + pgate[bh * Cc + c]));  // 128^-(0.5+p)
    float v[16];
    float mx = -3.4e38f;
    #pragma unroll
    for (int k = 0; k < 16; ++k) {
      v[k] = sb[c * Cc + lane + 32 * k] * invs;
      mx = fmaxf(mx, v[k]);
    }
    #pragma unroll
    for (int o = 16; o > 0; o >>= 1) mx = fmaxf(mx, __shfl_xor(mx, o, 32));
    float sum = 0.f;
    #pragma unroll
    for (int k = 0; k < 16; ++k) { v[k] = __expf(v[k] - mx); sum += v[k]; }
    #pragma unroll
    for (int o = 16; o > 0; o >>= 1) sum += __shfl_xor(sum, o, 32);
    float rs = 1.f / sum;
    #pragma unroll
    for (int k = 0; k < 16; ++k) wlds[ri * 520 + lane + 32 * k] = f2bf(v[k] * rs);
  }
  __syncthreads();
  int k0 = (lane >> 4) * 8;
  int d = wave * 16 + (lane & 15);                    // N col (head dim)
  const u16* vcol = V + b * PB + (h * Dd + d) * Cc;   // VT[b,h,d,e], e-contiguous
  v8f acc = {};
  for (int eb = 0; eb < 512; eb += 32) {
    v16bf A  = joinbf(ld8(&wlds[(lane & 15) * 520 + eb + k0]),
                      ld8(&wlds[(lane & 15) * 520 + eb + k0 + 16]));
    v16bf Bm = joinbf(ld8(vcol + eb + k0), ld8(vcol + eb + k0 + 16));
    acc = wmma_bf16(A, Bm, acc);
  }
  float wf = 1.f + (float)weight[0];
  #pragma unroll
  for (int g = 0; g < 8; ++g) {
    int c = c0 + g + 8 * (lane >> 4);
    int idx = b * PB + (h * Dd + d) * Cc + c;
    out[idx] = qpf[idx] + acc[g] * wf;
  }
}

// ---------------------------------------------------------------- launch
extern "C" void kernel_launch(void* const* d_in, const int* in_sizes, int n_in,
                              void* d_out, int out_size, void* d_ws, size_t ws_size,
                              hipStream_t stream) {
  (void)in_sizes; (void)n_in; (void)out_size; (void)ws_size;
  const float* x    = (const float*)d_in[0];
  const float* Wqk  = (const float*)d_in[1];
  const float* bqk  = (const float*)d_in[2];
  const float* Wp   = (const float*)d_in[3];
  const float* bp   = (const float*)d_in[4];
  const float* Wv   = (const float*)d_in[5];
  const float* bv   = (const float*)d_in[6];
  const int*   wgt  = (const int*)d_in[7];
  float* out = (float*)d_out;

  char* ws = (char*)d_ws;
  size_t off = 0;
  float* qpf = (float*)(ws + off); off += (size_t)Bn * PB * 4;   // avg pool f32
  u16* qpb   = (u16*)(ws + off);   off += (size_t)Bn * PB * 2;   // avg pool bf16
  u16* kpb   = (u16*)(ws + off);   off += (size_t)Bn * PB * 2;   // max pool bf16
  u16* wqkb  = (u16*)(ws + off);   off += (size_t)HNh * Dd * Dd * 2;
  u16* wvb   = (u16*)(ws + off);   off += (size_t)Cc * Cc * 2;
  u16* Qb    = (u16*)(ws + off);   off += (size_t)Bn * PB * 2;
  u16* Kb    = (u16*)(ws + off);   off += (size_t)Bn * PB * 2;
  u16* Vb    = (u16*)(ws + off);   off += (size_t)Bn * PB * 2;
  float* Sb  = (float*)(ws + off); off += (size_t)Bn * HNh * Cc * Cc * 4;
  float* pg  = (float*)(ws + off); off += (size_t)Bn * HNh * Cc * 4;

  pool_kernel  <<<Bn * PB / 256, 256, 0, stream>>>(x, qpf, qpb, kpb);
  cvt_kernel   <<<1536, 256, 0, stream>>>(Wqk, Wv, wqkb, wvb);
  qkproj_kernel<<<dim3(4, HNh, Bn), 256, 0, stream>>>(qpb, wqkb, bqk, Qb);
  qkproj_kernel<<<dim3(4, HNh, Bn), 256, 0, stream>>>(kpb, wqkb, bqk, Kb);
  vgemm_kernel <<<dim3(8, 4, Bn),   256, 0, stream>>>(qpb, wvb, bv, Vb);
  scores_kernel<<<dim3(4, 8, Bn * HNh), 256, 0, stream>>>(Qb, Kb, Sb);
  gate_kernel  <<<Bn * HNh, 256, 0, stream>>>(Sb, Wp, bp, pg);
  attn_kernel  <<<dim3(32, HNh, Bn), 256, 0, stream>>>(Sb, pg, Vb, qpf, wgt, out);
}
